// CTMActorCritic_66554813219392
// MI455X (gfx1250) — matrix operand order repaired and checked by reading, not compiled, not run on hardware
//
#include <hip/hip_runtime.h>
#include <hip/hip_bf16.h>
#include <math.h>

// ---------------- problem constants ----------------
#define BATCH   2048
#define OBS_D   256
#define ACT_D   64
#define DN      512       // D
#define SYN_H   2048
#define MH      8         // M (pre-history depth)
#define NLM_H   32
#define NSYNC   256
#define SWIN    8
#define NTICKS  8
#define INH     512
#define XDIM    (DN + INH)   // 1024

// d_out layout (floats): logits | values | pre_h_f | hist
#define OUT_LOGITS   0
#define OUT_VALUES   (BATCH * ACT_D)                        // 131072
#define OUT_PREH     (OUT_VALUES + BATCH)                   // 133120
#define OUT_HIST     (OUT_PREH + BATCH * DN * MH)           // 8521728

// d_ws layout (bf16 halfs)
#define WS_BBW1  0
#define WS_BBW2  (WS_BBW1 + INH * OBS_D)                    // 131072
#define WS_SW1   (WS_BBW2 + INH * INH)                      // 393216
#define WS_SW2   (WS_SW1  + SYN_H * XDIM)                   // 2490368
#define WS_SW3   (WS_SW2  + SYN_H * SYN_H)                  // 6684672
#define WS_OE    (WS_SW3  + DN * SYN_H)                     // 7733248
#define WS_END   (WS_OE   + BATCH * INH)

typedef __attribute__((ext_vector_type(16))) __bf16   v16bf;
typedef __attribute__((ext_vector_type(8)))  float    v8f;
typedef __attribute__((ext_vector_type(4)))  unsigned u32x4;
typedef __attribute__((ext_vector_type(4)))  float    f32x4;

union FragU { v16bf v; u32x4 q[2]; };

__device__ __forceinline__ __bf16 f2bf(float f) {
    union { float f; unsigned u; } x; x.f = f;
    unsigned r = x.u + 0x7FFFu + ((x.u >> 16) & 1u);   // round-to-nearest-even
    unsigned short h = (unsigned short)(r >> 16);
    __bf16 o; __builtin_memcpy(&o, &h, 2);
    return o;
}

__device__ __forceinline__ float gelu_exact(float x) {
    return 0.5f * x * (1.0f + erff(x * 0.70710678118654752f));
}

__device__ __forceinline__ v8f wmma_bf16(v16bf a, v16bf b, v8f c) {
    // D = A(16x32 bf16) * B(32x16 bf16) + C(16x16 f32)
    return __builtin_amdgcn_wmma_f32_16x16x32_bf16(
        /*neg_a=*/false, a, /*neg_b=*/false, b,
        /*c_mod=*/(short)0, c, /*reuse_a=*/false, /*reuse_b=*/false);
}

// A fragment: pa = lane base (row m, + lane-half k offset). 16-bit A 16x32 layout:
// lane<16: K = kt+[0..7], kt+[16..23];  lane>=16: K = kt+8+[0..7], kt+24+[0..7]
__device__ __forceinline__ v16bf load_a_frag(const __bf16* pa, int kt) {
    FragU u;
    u.q[0] = *(const u32x4*)(pa + kt);        // e0..7
    u.q[1] = *(const u32x4*)(pa + kt + 16);   // e8..15
    return u.v;
}

// B fragment: pb = lane base (weight row n, + lane-half k offset); same K layout as A.
__device__ __forceinline__ v16bf load_b_frag(const __bf16* pb, int kt) {
    FragU u;
    u.q[0] = *(const u32x4*)(pb + kt);
    u.q[1] = *(const u32x4*)(pb + kt + 16);
    return u.v;
}

// Dual-n-tile GEMM: one A fragment (LDS) feeds TWO adjacent 16x16 output tiles,
// halving ds_read traffic per v_wmma. Unroll capped at 4 to avoid VGPR spills.
__device__ __forceinline__ void gemm_acc2(const __bf16* A, int lda,
                                          const __bf16* W, int K,
                                          const float* bias, int nt0, int lane,
                                          v8f& acc0, v8f& acc1) {
    const int nlane = lane & 15;
    float bA = bias[nt0 * 16 + nlane];
    float bB = bias[(nt0 + 1) * 16 + nlane];
#pragma unroll
    for (int i = 0; i < 8; ++i) { acc0[i] = bA; acc1[i] = bB; }

    const __bf16* pa  = A + nlane * lda + ((lane & 16) ? 8 : 0);
    const __bf16* pb0 = W + (size_t)(nt0 * 16 + nlane) * K + ((lane & 16) ? 8 : 0);
    const __bf16* pb1 = pb0 + (size_t)16 * K;   // row n+16 == next n-tile, same lane

#pragma unroll 4
    for (int kt = 0; kt < K; kt += 32) {
        v16bf a  = load_a_frag(pa, kt);
        v16bf b0 = load_b_frag(pb0, kt);
        v16bf b1 = load_b_frag(pb1, kt);
        if (kt + 256 < K) {
            __builtin_prefetch(pb0 + kt + 256, 0, 0);   // global_prefetch_b8
            __builtin_prefetch(pb1 + kt + 256, 0, 0);
        }
        acc0 = wmma_bf16(a, b0, acc0);
        acc1 = wmma_bf16(a, b1, acc1);
    }
}

// D-fragment stores. 32-bit C/D 16x16: VGPR v -> M = v (+8 for lanes>=16), N = lane&15.
__device__ __forceinline__ void store_d_lds_bf16(__bf16* dst, int ldd, int ntile, int lane,
                                                 v8f acc, bool act) {
    int n  = ntile * 16 + (lane & 15);
    int mb = (lane & 16) ? 8 : 0;
#pragma unroll
    for (int v = 0; v < 8; ++v) {
        float x = acc[v];
        if (act) x = gelu_exact(x);
        dst[(mb + v) * ldd + n] = f2bf(x);
    }
}

__device__ __forceinline__ void store_d_lds_f32(float* dst, int ldd, int ntile, int lane, v8f acc) {
    int n  = ntile * 16 + (lane & 15);
    int mb = (lane & 16) ? 8 : 0;
#pragma unroll
    for (int v = 0; v < 8; ++v) dst[(mb + v) * ldd + n] = acc[v];
}

__device__ __forceinline__ void store_d_gbl_bf16(__bf16* dst, int ldd, int ntile, int lane, v8f acc) {
    int n  = ntile * 16 + (lane & 15);
    int mb = (lane & 16) ? 8 : 0;
#pragma unroll
    for (int v = 0; v < 8; ++v) dst[(size_t)(mb + v) * ldd + n] = f2bf(acc[v]);
}

// ---------------- kernel 0: f32 -> bf16 weight conversion ----------------
__global__ void cvt_bf16_kernel(const float* __restrict__ s, __bf16* __restrict__ d, int n) {
    for (int i = blockIdx.x * blockDim.x + threadIdx.x; i < n; i += gridDim.x * blockDim.x)
        d[i] = f2bf(s[i]);
}

// ---------------- kernel 1: backbone obs_embed (B,INH) as bf16 ----------------
// obs_embed = Lin2( gelu( Lin1(obs) ) ); 16-row batch tile per block, 8 waves.
__global__ void __launch_bounds__(256, 1)
backbone_kernel(const float* __restrict__ obs,
                const __bf16* __restrict__ bw1, const float* __restrict__ bb1,
                const __bf16* __restrict__ bw2, const float* __restrict__ bb2,
                __bf16* __restrict__ oe) {
    __shared__ __bf16 xo[16 * OBS_D];   // 8 KB
    __shared__ __bf16 t1[16 * INH];     // 16 KB
    const int tid = threadIdx.x, lane = tid & 31, wv = tid >> 5;
    const int b0 = blockIdx.x * 16;

    for (int i = tid; i < 16 * OBS_D; i += 256) {
        int r = i >> 8, c = i & 255;
        xo[r * OBS_D + c] = f2bf(obs[(size_t)(b0 + r) * OBS_D + c]);
    }
    __syncthreads();
    for (int nt = wv * 2; nt < INH / 16; nt += 16) {
        v8f a0, a1;
        gemm_acc2(xo, OBS_D, bw1, OBS_D, bb1, nt, lane, a0, a1);
        store_d_lds_bf16(t1, INH, nt, lane, a0, true);
        store_d_lds_bf16(t1, INH, nt + 1, lane, a1, true);
    }
    __syncthreads();
    __bf16* oeT = oe + (size_t)b0 * INH;
    for (int nt = wv * 2; nt < INH / 16; nt += 16) {
        v8f a0, a1;
        gemm_acc2(t1, INH, bw2, INH, bb2, nt, lane, a0, a1);
        store_d_gbl_bf16(oeT, INH, nt, lane, a0);       // no activation
        store_d_gbl_bf16(oeT, INH, nt + 1, lane, a1);
    }
}

// ---------------- kernel 2: fused 8-tick recurrent core ----------------
// One block = 16 batch rows, 8 waves, 192KB dynamic LDS:
//   xbuf (16x1024 bf16) | a1 (16x2048 bf16) | a2 (16x2048 bf16) | preb (16x512 f32)
__global__ void __launch_bounds__(256, 1)
tick_kernel(const float* __restrict__ pre_h_in,
            const float* __restrict__ post_hist,
            const __bf16* __restrict__ oe,
            const __bf16* __restrict__ w1, const float* __restrict__ b1,
            const __bf16* __restrict__ w2, const float* __restrict__ b2,
            const __bf16* __restrict__ w3, const float* __restrict__ b3,
            const float* __restrict__ nw1, const float* __restrict__ nb1,
            const float* __restrict__ nwo, const float* __restrict__ nbo,
            float* __restrict__ ph_out, float* __restrict__ hist_out) {
    extern __shared__ char smem[];
    __bf16* xbuf = (__bf16*)smem;                                  // 32768 B
    __bf16* a1   = (__bf16*)(smem + 32768);                        // 65536 B
    __bf16* a2   = (__bf16*)(smem + 32768 + 65536);                // 65536 B
    float*  preb = (float*)(smem + 32768 + 65536 + 65536);         // 32768 B

    const int tid = threadIdx.x, lane = tid & 31, wv = tid >> 5;
    const int b0 = blockIdx.x * 16;

    // init x = [post0 (bf16) | obs_embed (bf16)]
    for (int i = tid; i < 16 * DN; i += 256) {
        int r = i >> 9, d = i & 511;
        xbuf[r * XDIM + d] = f2bf(post_hist[(size_t)(b0 + r) * DN + d]);
    }
    for (int i = tid; i < 16 * (INH / 8); i += 256) {   // 8-half chunks
        int r = i >> 6, c8 = (i & 63) << 3;
        *(u32x4*)(xbuf + r * XDIM + DN + c8) =
            *(const u32x4*)(oe + (size_t)(b0 + r) * INH + c8);
    }

    for (int t = 0; t < NTICKS; ++t) {
        __syncthreads();
        // layer 1: a1 = gelu(x @ w1^T + b1)      (K=1024, N=2048)
        for (int nt = wv * 2; nt < SYN_H / 16; nt += 16) {
            v8f c0, c1;
            gemm_acc2(xbuf, XDIM, w1, XDIM, b1, nt, lane, c0, c1);
            store_d_lds_bf16(a1, SYN_H, nt, lane, c0, true);
            store_d_lds_bf16(a1, SYN_H, nt + 1, lane, c1, true);
        }
        __syncthreads();
        // layer 2: a2 = gelu(a1 @ w2^T + b2)     (K=2048, N=2048)
        for (int nt = wv * 2; nt < SYN_H / 16; nt += 16) {
            v8f c0, c1;
            gemm_acc2(a1, SYN_H, w2, SYN_H, b2, nt, lane, c0, c1);
            store_d_lds_bf16(a2, SYN_H, nt, lane, c0, true);
            store_d_lds_bf16(a2, SYN_H, nt + 1, lane, c1, true);
        }
        __syncthreads();
        // layer 3: pre = a2 @ w3^T + b3          (K=2048, N=512, f32)
        for (int nt = wv * 2; nt < DN / 16; nt += 16) {
            v8f c0, c1;
            gemm_acc2(a2, SYN_H, w3, SYN_H, b3, nt, lane, c0, c1);
            store_d_lds_f32(preb, DN, nt, lane, c0);
            store_d_lds_f32(preb, DN, nt + 1, lane, c1);
        }
        __syncthreads();

        // FIFO + per-neuron grouped MLP; thread -> (batch row, 32 consecutive d)
        const int bloc = tid & 15;
        const int dbase = (tid >> 4) * 32;
        const int b = b0 + bloc;
        for (int dd = 0; dd < 32; ++dd) {
            const int d = dbase + dd;
            const size_t phoff = ((size_t)b * DN + d) * MH;
            const float* src = (t == 0) ? (pre_h_in + phoff) : (ph_out + phoff);
            f32x4 A0 = ((const f32x4*)src)[0];
            f32x4 A1 = ((const f32x4*)src)[1];
            float pre = preb[bloc * DN + d];
            float nh[8] = {A0.y, A0.z, A0.w, A1.x, A1.y, A1.z, A1.w, pre};
            float o = nbo[d];
            for (int j = 0; j < NLM_H; ++j) {
                const float* wr = nw1 + ((size_t)d * NLM_H + j) * MH;
                float s = nb1[d * NLM_H + j];
#pragma unroll
                for (int m = 0; m < MH; ++m) s += nh[m] * wr[m];
                o += gelu_exact(s) * nwo[d * NLM_H + j];
            }
            float post = tanhf(o);
            f32x4* dst4 = (f32x4*)(ph_out + phoff);
            dst4[0] = (f32x4){nh[0], nh[1], nh[2], nh[3]};
            dst4[1] = (f32x4){nh[4], nh[5], nh[6], nh[7]};
            hist_out[phoff + t] = post;              // hist (B,D,8) shares stride w/ pre_h
            xbuf[bloc * XDIM + d] = f2bf(post);      // next tick's input
        }
    }
}

// ---------------- kernel 3: sync pairs + proj + actor/critic heads ----------------
// PAIRS collapse to r=0, c = (p<128 ? p : 256+p). weighted dot over SWIN with decay^2.
__global__ void heads_kernel(const float* __restrict__ hist,
                             const float* __restrict__ proj_w, const float* __restrict__ proj_b,
                             const float* __restrict__ actor_w, const float* __restrict__ actor_b,
                             const float* __restrict__ critic_w, const float* __restrict__ critic_b,
                             float* __restrict__ logits, float* __restrict__ values) {
    __shared__ float sraw[NSYNC];
    __shared__ float ssync[NSYNC];
    __shared__ float h0s[SWIN];
    const int tid = threadIdx.x;
    const int b = blockIdx.x;

    if (tid < SWIN) h0s[tid] = hist[((size_t)b * DN + 0) * SWIN + tid];
    __syncthreads();

    float d2[SWIN];
    { float v = 1.0f;
      for (int i = SWIN - 1; i >= 0; --i) { d2[i] = v; v *= 0.81f; } }  // (0.9^(7-i))^2

    const int c = (tid < 128) ? tid : (256 + tid);
    const float* hc = hist + ((size_t)b * DN + c) * SWIN;
    float s = 0.0f;
#pragma unroll
    for (int t = 0; t < SWIN; ++t) s += h0s[t] * hc[t] * d2[t];
    sraw[tid] = s;
    __syncthreads();

    float acc = proj_b[tid];
    const float* pw = proj_w + (size_t)tid * NSYNC;
    for (int q = 0; q < NSYNC; ++q) acc += sraw[q] * pw[q];
    ssync[tid] = acc;
    __syncthreads();

    if (tid < ACT_D) {
        float a = actor_b[tid];
        const float* aw = actor_w + (size_t)tid * NSYNC;
        for (int q = 0; q < NSYNC; ++q) a += ssync[q] * aw[q];
        logits[(size_t)b * ACT_D + tid] = a;
    }
    if (tid == ACT_D) {
        float v = critic_b[0];
        for (int q = 0; q < NSYNC; ++q) v += ssync[q] * critic_w[q];
        values[b] = v;
    }
}

// ---------------- host launcher ----------------
extern "C" void kernel_launch(void* const* d_in, const int* in_sizes, int n_in,
                              void* d_out, int out_size, void* d_ws, size_t ws_size,
                              hipStream_t stream) {
    const float* obs       = (const float*)d_in[0];
    const float* pre_h     = (const float*)d_in[1];
    const float* post_hist = (const float*)d_in[2];
    const float* bb_w1     = (const float*)d_in[3];
    const float* bb_b1     = (const float*)d_in[4];
    const float* bb_w2     = (const float*)d_in[5];
    const float* bb_b2     = (const float*)d_in[6];
    const float* syn_w1    = (const float*)d_in[7];
    const float* syn_b1    = (const float*)d_in[8];
    const float* syn_w2    = (const float*)d_in[9];
    const float* syn_b2    = (const float*)d_in[10];
    const float* syn_w3    = (const float*)d_in[11];
    const float* syn_b3    = (const float*)d_in[12];
    const float* nlm_w1    = (const float*)d_in[13];
    const float* nlm_b1    = (const float*)d_in[14];
    const float* nlm_wout  = (const float*)d_in[15];
    const float* nlm_bout  = (const float*)d_in[16];
    const float* proj_w    = (const float*)d_in[17];
    const float* proj_b    = (const float*)d_in[18];
    const float* actor_w   = (const float*)d_in[19];
    const float* actor_b   = (const float*)d_in[20];
    const float* critic_w  = (const float*)d_in[21];
    const float* critic_b  = (const float*)d_in[22];

    float* out     = (float*)d_out;
    float* logits  = out + OUT_LOGITS;
    float* values  = out + OUT_VALUES;
    float* ph_out  = out + OUT_PREH;
    float* hist    = out + OUT_HIST;

    __bf16* ws = (__bf16*)d_ws;

    // 0) bf16 weight conversion (weights then live in L2 for the whole call)
    cvt_bf16_kernel<<<512, 256, 0, stream>>>(bb_w1,  ws + WS_BBW1, INH * OBS_D);
    cvt_bf16_kernel<<<512, 256, 0, stream>>>(bb_w2,  ws + WS_BBW2, INH * INH);
    cvt_bf16_kernel<<<1024, 256, 0, stream>>>(syn_w1, ws + WS_SW1, SYN_H * XDIM);
    cvt_bf16_kernel<<<1024, 256, 0, stream>>>(syn_w2, ws + WS_SW2, SYN_H * SYN_H);
    cvt_bf16_kernel<<<1024, 256, 0, stream>>>(syn_w3, ws + WS_SW3, DN * SYN_H);

    // 1) backbone -> obs_embed (bf16)
    backbone_kernel<<<BATCH / 16, 256, 0, stream>>>(
        obs, ws + WS_BBW1, bb_b1, ws + WS_BBW2, bb_b2, ws + WS_OE);

    // 2) fused 8-tick recurrent core (192 KB dynamic LDS of the WGP's 320 KB)
    const int smem_bytes = 32768 + 65536 + 65536 + 32768;
    (void)hipFuncSetAttribute((const void*)tick_kernel,
                              hipFuncAttributeMaxDynamicSharedMemorySize, smem_bytes);
    tick_kernel<<<BATCH / 16, 256, smem_bytes, stream>>>(
        pre_h, post_hist, ws + WS_OE,
        ws + WS_SW1, syn_b1, ws + WS_SW2, syn_b2, ws + WS_SW3, syn_b3,
        nlm_w1, nlm_b1, nlm_wout, nlm_bout,
        ph_out, hist);

    // 3) synchronisation features + heads
    heads_kernel<<<BATCH, 256, 0, stream>>>(
        hist, proj_w, proj_b, actor_w, actor_b, critic_w, critic_b, logits, values);
}
